// ImprovedAttentionModel_77438260347428
// MI455X (gfx1250) — compile-verified, matrix operand order checked by reading, not run on hardware
//
#include <hip/hip_runtime.h>
#include <hip/hip_bf16.h>

// ---------------------------------------------------------------------------
// Problem constants
// ---------------------------------------------------------------------------
constexpr int N_  = 8192;   // rows
constexpr int D_  = 1024;   // feature dim
constexpr int C_  = 1000;   // classes
constexpr int CP_ = 1024;   // C padded to multiple of 64 (zero-filled)

typedef __attribute__((ext_vector_type(16))) __bf16 v16bf;
typedef __attribute__((ext_vector_type(8)))  float  v8f;
typedef __attribute__((ext_vector_type(4)))  unsigned int u32x4;
typedef __attribute__((ext_vector_type(8)))  int i32x8;
typedef __attribute__((ext_vector_type(4)))  int i32x4;

// Scheduling-group helper (mask: 0x8 = MFMA/WMMA, 0x20 = VMEM read).
#if defined(__has_builtin)
#if __has_builtin(__builtin_amdgcn_sched_group_barrier)
#define SGB(MASK, SIZE, ID) __builtin_amdgcn_sched_group_barrier(MASK, SIZE, ID)
#else
#define SGB(MASK, SIZE, ID)
#endif
#else
#define SGB(MASK, SIZE, ID)
#endif

// ---------------------------------------------------------------------------
// fp32 -> bf16 flat conversion
// ---------------------------------------------------------------------------
__global__ void cvt_bf16_kernel(const float* __restrict__ in,
                                __bf16* __restrict__ out, size_t n) {
    size_t i = (size_t)blockIdx.x * blockDim.x + threadIdx.x;
    if (i < n) out[i] = (__bf16)in[i];
}

// ---------------------------------------------------------------------------
// Transpose + convert: in is [rows][cols] fp32, out is [padded_cols][rows] bf16
// ---------------------------------------------------------------------------
__global__ void transpose_cvt_kernel(const float* __restrict__ in,
                                     __bf16* __restrict__ out,
                                     int rows, int cols, int padded_cols) {
    size_t idx = (size_t)blockIdx.x * blockDim.x + threadIdx.x;
    size_t total = (size_t)padded_cols * rows;
    if (idx >= total) return;
    int j = (int)(idx / rows);
    int i = (int)(idx % rows);
    float v = (j < cols) ? in[(size_t)i * cols + j] : 0.0f;
    out[idx] = (__bf16)v;
}

// ---------------------------------------------------------------------------
// Projection GEMM: out = X[N,D] @ W[D,D] + bias, using Wt = W^T (bf16).
// One wave computes a 16x64 tile; sched-group barriers pin a one-chunk
// lookahead pipeline (loads of chunk c+1 in flight over chunk c's WMMAs).
// MODE 0: bf16 [N][D]; MODE 1: bf16 transposed [D][N]; MODE 2: fp32 [N][D]
// ---------------------------------------------------------------------------
template <int MODE>
__global__ void proj_gemm_kernel(const __bf16* __restrict__ X,
                                 const __bf16* __restrict__ Wt,
                                 const float* __restrict__ bias,
                                 void* __restrict__ outp) {
    const int lane = threadIdx.x & 31;
    const int half = lane >> 4;
    const int ml   = lane & 15;
    const int row0 = blockIdx.x * 16;
    const int col0 = blockIdx.y * 64;

    const __bf16* xp = X  + (size_t)(row0 + ml) * D_ + half * 16;
    const __bf16* wp = Wt + (size_t)(col0 + ml) * D_ + half * 16;

    float bvs[4];
#pragma unroll
    for (int j = 0; j < 4; ++j) bvs[j] = bias[col0 + j * 16 + ml];

    v8f acc[4];
#pragma unroll
    for (int j = 0; j < 4; ++j) acc[j] = (v8f){};

    v16bf a[2];
    v16bf b[2][4];
    a[0] = *(const v16bf*)xp;
#pragma unroll
    for (int j = 0; j < 4; ++j)
        b[0][j] = *(const v16bf*)(wp + (size_t)j * 16 * D_);

#pragma unroll
    for (int c = 0; c < D_ / 32; ++c) {
        const int cur = c & 1, nxt = cur ^ 1;
        if (c < D_ / 32 - 1) {
            a[nxt] = *(const v16bf*)(xp + (c + 1) * 32);
#pragma unroll
            for (int j = 0; j < 4; ++j)
                b[nxt][j] = *(const v16bf*)(wp + (size_t)j * 16 * D_ + (c + 1) * 32);
        }
#pragma unroll
        for (int j = 0; j < 4; ++j)
            acc[j] = __builtin_amdgcn_wmma_f32_16x16x32_bf16(
                false, a[cur], false, b[cur][j], (short)0, acc[j], false, false);
    }
    // Pipeline shape: [bias + chunk0 + chunk1 loads][4 WMMA][10 loads]...[8 WMMA]
    SGB(0x020, 24, 0);
#pragma unroll
    for (int it = 0; it < 30; ++it) {
        SGB(0x008, 4, 0);
        SGB(0x020, 10, 0);
    }
    SGB(0x008, 8, 0);

#pragma unroll
    for (int j = 0; j < 4; ++j) {
        const int col = col0 + j * 16 + ml;
        if constexpr (MODE == 0) {
            __bf16* o = (__bf16*)outp;
#pragma unroll
            for (int r = 0; r < 8; ++r)
                o[(size_t)(row0 + r + 8 * half) * D_ + col] = (__bf16)(acc[j][r] + bvs[j]);
        } else if constexpr (MODE == 1) {
            __bf16* o = (__bf16*)outp;
#pragma unroll
            for (int r = 0; r < 8; ++r)
                o[(size_t)col * N_ + row0 + r + 8 * half] = (__bf16)(acc[j][r] + bvs[j]);
        } else {
            float* o = (float*)outp;
#pragma unroll
            for (int r = 0; r < 8; ++r)
                o[(size_t)(row0 + r + 8 * half) * D_ + col] = acc[j][r] + bvs[j];
        }
    }
}

// ---------------------------------------------------------------------------
// Flash attention: A = softmax(Q K^T / 32) V, outputs fp32 [N][D].
// Grid: N/16 blocks of 256 threads (8 waves). Block owns 16 query rows.
// Wave w owns D-slice [128w, 128w+128): Q frags + O accumulator in VGPRs.
// S is 8-way split-K reduced through LDS (b128 ops); softmax state (m,l)
// maintained redundantly per wave; P round-trips through per-wave LDS.
// Sched-group barriers pin a one-tile lookahead pipeline in both WMMA phases.
// Wave 0 drives the Tensor Data Mover to pull the next key block toward
// LDS/L2 (prefetch-only; consumed data flows through the normal load path).
// ---------------------------------------------------------------------------
__global__ void __launch_bounds__(256)
flash_attn_kernel(const __bf16* __restrict__ Q, const __bf16* __restrict__ K,
                  const __bf16* __restrict__ Vt, float* __restrict__ Aout) {
    const int tid  = threadIdx.x;
    const int wave = tid >> 5;
    const int lane = tid & 31;
    const int half = lane >> 4;
    const int ml   = lane & 15;
    const int q0   = blockIdx.x * 16;
    const int d0   = wave * 128;

    __shared__ __align__(32) float  s_red[8][4][32][8];   // 32 KB split-K S partials
    __shared__ __align__(32) __bf16 s_p[8][16][64];       // 16 KB per-wave P staging
    __shared__ __align__(256) char  s_tdm[16384];         // 16 KB TDM landing pad

    // Resident Q fragments for this wave's 128-wide D-slice: 4 chunks of K=32.
    v16bf aq[4];
#pragma unroll
    for (int c = 0; c < 4; ++c)
        aq[c] = *(const v16bf*)(Q + (size_t)(q0 + ml) * D_ + d0 + c * 32 + half * 16);

    v8f o[8];
#pragma unroll
    for (int dt = 0; dt < 8; ++dt) o[dt] = (v8f){};
    float m[8], l[8];
#pragma unroll
    for (int r = 0; r < 8; ++r) { m[r] = -3.0e38f; l[r] = 0.0f; }

    for (int j0 = 0; j0 < N_; j0 += 64) {
        // --- TDM prefetch of the next 64x64-element corner of the K block ---
        if (wave == 0 && j0 + 64 < N_) {
            unsigned lds_off = (unsigned)(size_t)&s_tdm[0];
            unsigned long long ga =
                (unsigned long long)(const void*)(K + (size_t)(j0 + 64) * D_);
            u32x4 g0;
            g0[0] = 1u;                                   // count=1, user D#
            g0[1] = lds_off;                              // lds_addr
            g0[2] = (unsigned)(ga & 0xffffffffu);         // global_addr[31:0]
            g0[3] = (unsigned)((ga >> 32) & 0x01ffffffu)  // global_addr[56:32]
                    | (2u << 30);                         // type=2 (image)
            i32x8 g1;
            g1[0] = 1 << 16;                 // workgroup_mask=0, data_size=2B
            g1[1] = (int)(1024u << 16);      // tensor_dim0 = 1024
            g1[2] = (int)(8192u << 16);      // tensor_dim1 = 8192
            g1[3] = (int)(64u << 16);        // tile_dim0 = 64 elements
            g1[4] = 64;                      // tile_dim1 = 64 rows
            g1[5] = 1024;                    // tensor_dim0_stride = 1024
            g1[6] = 0;
            g1[7] = 0;
            i32x4 gz4 = (i32x4)0;
            i32x8 gz8 = (i32x8)0;
            __builtin_amdgcn_tensor_load_to_lds(g0, g1, gz4, gz4, gz8, 0);
        }

        // ---- partial S over this wave's 128-wide D slice: 4 key tiles of
        //      4 chunks each; one tile of loads kept in flight over WMMAs ----
        const __bf16* kbase = K + (size_t)(j0 + ml) * D_ + d0 + half * 16;
        v16bf kb[2][4];
#pragma unroll
        for (int i = 0; i < 4; ++i)
            kb[0][i] = *(const v16bf*)(kbase + i * 32);
#pragma unroll
        for (int t = 0; t < 4; ++t) {
            if (t < 3) {
#pragma unroll
                for (int i = 0; i < 4; ++i)
                    kb[(t + 1) & 1][i] =
                        *(const v16bf*)(kbase + (size_t)(t + 1) * 16 * D_ + i * 32);
            }
            v8f sp = {};
#pragma unroll
            for (int i = 0; i < 4; ++i)
                sp = __builtin_amdgcn_wmma_f32_16x16x32_bf16(
                    false, aq[i], false, kb[t & 1][i], (short)0, sp, false, false);
            *(v8f*)&s_red[wave][t][lane][0] = sp;
        }
        // Pipeline: [16 loads][4 WMMA][8 loads][4 WMMA][8 loads][4 WMMA][4 WMMA]
        SGB(0x020, 16, 0);
        SGB(0x008, 4, 0);
        SGB(0x020, 8, 0);
        SGB(0x008, 4, 0);
        SGB(0x020, 8, 0);
        SGB(0x008, 8, 0);
        __syncthreads();

        v8f s[4];
#pragma unroll
        for (int t = 0; t < 4; ++t) {
            v8f acc = *(const v8f*)&s_red[0][t][lane][0];
#pragma unroll
            for (int w = 1; w < 8; ++w)
                acc += *(const v8f*)&s_red[w][t][lane][0];
            s[t] = acc * 0.03125f;   // 1/sqrt(1024)
        }
        __syncthreads();

        // ---- online softmax update (rows live in C-layout: N == lane&15) ----
#pragma unroll
        for (int r = 0; r < 8; ++r) {
            float mx = fmaxf(fmaxf(s[0][r], s[1][r]), fmaxf(s[2][r], s[3][r]));
            mx = fmaxf(mx, __shfl_xor(mx, 1, 32));
            mx = fmaxf(mx, __shfl_xor(mx, 2, 32));
            mx = fmaxf(mx, __shfl_xor(mx, 4, 32));
            mx = fmaxf(mx, __shfl_xor(mx, 8, 32));
            float mn = fmaxf(m[r], mx);
            float al = __expf(m[r] - mn);
            float rs = 0.0f;
#pragma unroll
            for (int t = 0; t < 4; ++t) {
                float p = __expf(s[t][r] - mn);
                s[t][r] = p;
                rs += p;
            }
            rs += __shfl_xor(rs, 1, 32);
            rs += __shfl_xor(rs, 2, 32);
            rs += __shfl_xor(rs, 4, 32);
            rs += __shfl_xor(rs, 8, 32);
            l[r] = l[r] * al + rs;
            m[r] = mn;
#pragma unroll
            for (int t = 0; t < 4; ++t)
                s_p[wave][r + 8 * half][t * 16 + ml] = (__bf16)s[t][r];
#pragma unroll
            for (int dt = 0; dt < 8; ++dt) o[dt][r] *= al;
        }

        // ---- O += P @ V for this wave's D slice: 4 groups of 4 d-tiles;
        //      first 16 V loads hoisted over the softmax VALU work ----
        v16bf pa[2];
#pragma unroll
        for (int kc = 0; kc < 2; ++kc)
            pa[kc] = *(const v16bf*)&s_p[wave][ml][kc * 32 + half * 16];

        const __bf16* vbase = Vt + (size_t)(d0 + ml) * N_ + j0 + half * 16;
        v16bf vb[2][4];
#pragma unroll
        for (int i = 0; i < 4; ++i)
            vb[0][i] = *(const v16bf*)(vbase + (size_t)i * 16 * N_);
#pragma unroll
        for (int gg = 0; gg < 4; ++gg) {
            const int kc = gg >> 1, g = gg & 1;
            if (gg < 3) {
                const int gn = gg + 1;
                const int kcn = gn >> 1, gno = gn & 1;
#pragma unroll
                for (int i = 0; i < 4; ++i)
                    vb[gn & 1][i] = *(const v16bf*)(vbase +
                        (size_t)(gno * 4 + i) * 16 * N_ + kcn * 32);
            }
#pragma unroll
            for (int i = 0; i < 4; ++i) {
                const int dt = g * 4 + i;
                o[dt] = __builtin_amdgcn_wmma_f32_16x16x32_bf16(
                    false, pa[kc], false, vb[gg & 1][i], (short)0, o[dt],
                    false, false);
            }
        }
        // Pipeline: [16 loads][4 WMMA][8 loads][4 WMMA][8 loads][8 WMMA]
        SGB(0x020, 16, 0);
        SGB(0x008, 4, 0);
        SGB(0x020, 8, 0);
        SGB(0x008, 4, 0);
        SGB(0x020, 8, 0);
        SGB(0x008, 8, 0);
    }

    __builtin_amdgcn_s_wait_tensorcnt(0);

    // ---- epilogue: normalize and store fp32 ----
#pragma unroll
    for (int r = 0; r < 8; ++r) l[r] = 1.0f / l[r];
#pragma unroll
    for (int dt = 0; dt < 8; ++dt)
#pragma unroll
        for (int r = 0; r < 8; ++r)
            Aout[(size_t)(q0 + r + 8 * half) * D_ + d0 + dt * 16 + ml] =
                o[dt][r] * l[r];
}

// ---------------------------------------------------------------------------
// Gate + mix: beta = sigmoid([a, x_r, a-x_r] @ W_beta + b_beta) per row,
// mix = beta*x_r + (1-beta)*a, stored bf16 for the final GEMM.
// ---------------------------------------------------------------------------
__global__ void beta_mix_kernel(const float* __restrict__ a,
                                const float* __restrict__ xr,
                                const float* __restrict__ Wb,
                                const float* __restrict__ bb,
                                __bf16* __restrict__ mix) {
    const int n   = blockIdx.x;
    const int tid = threadIdx.x;
    float part = 0.0f;
    for (int d = tid; d < D_; d += 256) {
        float av = a[(size_t)n * D_ + d];
        float xv = xr[(size_t)n * D_ + d];
        part += av * Wb[d] + xv * Wb[D_ + d] + (av - xv) * Wb[2 * D_ + d];
    }
#pragma unroll
    for (int msk = 16; msk >= 1; msk >>= 1) part += __shfl_xor(part, msk, 32);
    __shared__ float red[8];
    if ((tid & 31) == 0) red[tid >> 5] = part;
    __syncthreads();
    float tot = 0.0f;
#pragma unroll
    for (int w = 0; w < 8; ++w) tot += red[w];
    const float beta = 1.0f / (1.0f + __expf(-(tot + bb[0])));
    for (int d = tid; d < D_; d += 256) {
        float av = a[(size_t)n * D_ + d];
        float xv = xr[(size_t)n * D_ + d];
        mix[(size_t)n * D_ + d] = (__bf16)(beta * xv + (1.0f - beta) * av);
    }
}

// ---------------------------------------------------------------------------
// Final GEMM: out[N,1000] = mix[N,1024] @ W_fc[1024,1000] + b_fc, fp32 out.
// 16x64 tile per wave from W_fc^T padded to 1024 rows (zeros beyond 1000).
// ---------------------------------------------------------------------------
__global__ void fc_gemm_kernel(const __bf16* __restrict__ X,
                               const __bf16* __restrict__ Wt,
                               const float* __restrict__ bias,
                               float* __restrict__ out) {
    const int lane = threadIdx.x & 31;
    const int half = lane >> 4;
    const int ml   = lane & 15;
    const int row0 = blockIdx.x * 16;
    const int col0 = blockIdx.y * 64;

    const __bf16* xp = X  + (size_t)(row0 + ml) * D_ + half * 16;
    const __bf16* wp = Wt + (size_t)(col0 + ml) * D_ + half * 16;

    v8f acc[4];
#pragma unroll
    for (int j = 0; j < 4; ++j) acc[j] = (v8f){};

    v16bf a[2];
    v16bf b[2][4];
    a[0] = *(const v16bf*)xp;
#pragma unroll
    for (int j = 0; j < 4; ++j)
        b[0][j] = *(const v16bf*)(wp + (size_t)j * 16 * D_);

#pragma unroll
    for (int c = 0; c < D_ / 32; ++c) {
        const int cur = c & 1, nxt = cur ^ 1;
        if (c < D_ / 32 - 1) {
            a[nxt] = *(const v16bf*)(xp + (c + 1) * 32);
#pragma unroll
            for (int j = 0; j < 4; ++j)
                b[nxt][j] = *(const v16bf*)(wp + (size_t)j * 16 * D_ + (c + 1) * 32);
        }
#pragma unroll
        for (int j = 0; j < 4; ++j)
            acc[j] = __builtin_amdgcn_wmma_f32_16x16x32_bf16(
                false, a[cur], false, b[cur][j], (short)0, acc[j], false, false);
    }
    // Pipeline shape: [chunk0 + chunk1 loads][4 WMMA][10 loads]...[8 WMMA]
    SGB(0x020, 20, 0);
#pragma unroll
    for (int it = 0; it < 30; ++it) {
        SGB(0x008, 4, 0);
        SGB(0x020, 10, 0);
    }
    SGB(0x008, 8, 0);

#pragma unroll
    for (int j = 0; j < 4; ++j) {
        const int col = col0 + j * 16 + ml;
        if (col < C_) {
            const float bvs = bias[col];
#pragma unroll
            for (int r = 0; r < 8; ++r)
                out[(size_t)(row0 + r + 8 * half) * C_ + col] = acc[j][r] + bvs;
        }
    }
}

// ---------------------------------------------------------------------------
// Host launcher
// ---------------------------------------------------------------------------
extern "C" void kernel_launch(void* const* d_in, const int* in_sizes, int n_in,
                              void* d_out, int out_size, void* d_ws, size_t ws_size,
                              hipStream_t stream) {
    const float* x     = (const float*)d_in[0];
    const float* Wskip = (const float*)d_in[1];
    const float* bskip = (const float*)d_in[2];
    const float* Wq    = (const float*)d_in[3];
    const float* bq    = (const float*)d_in[4];
    const float* Wk    = (const float*)d_in[5];
    const float* bk    = (const float*)d_in[6];
    const float* Wv    = (const float*)d_in[7];
    const float* bv    = (const float*)d_in[8];
    const float* Wbeta = (const float*)d_in[9];
    const float* bbeta = (const float*)d_in[10];
    const float* Wfc   = (const float*)d_in[11];
    const float* bfc   = (const float*)d_in[12];
    float* out = (float*)d_out;

    char* ws = (char*)d_ws;
    size_t off = 0;
    auto alloc = [&](size_t bytes) -> char* {
        char* p = ws + off;
        off += (bytes + 255) & ~(size_t)255;
        return p;
    };
    __bf16* xb   = (__bf16*)alloc((size_t)N_ * D_ * 2);
    __bf16* qb   = (__bf16*)alloc((size_t)N_ * D_ * 2);
    __bf16* kb   = (__bf16*)alloc((size_t)N_ * D_ * 2);
    __bf16* vTb  = (__bf16*)alloc((size_t)D_ * N_ * 2);
    __bf16* wqT  = (__bf16*)alloc((size_t)D_ * D_ * 2);
    __bf16* wkT  = (__bf16*)alloc((size_t)D_ * D_ * 2);
    __bf16* wvT  = (__bf16*)alloc((size_t)D_ * D_ * 2);
    __bf16* wsT  = (__bf16*)alloc((size_t)D_ * D_ * 2);
    __bf16* wfcT = (__bf16*)alloc((size_t)CP_ * D_ * 2);
    float*  xr   = (float*)alloc((size_t)N_ * D_ * 4);
    float*  av   = (float*)alloc((size_t)N_ * D_ * 4);
    __bf16* mix  = (__bf16*)alloc((size_t)N_ * D_ * 2);

    // --- stage 1: conversions ---
    {
        size_t n = (size_t)N_ * D_;
        cvt_bf16_kernel<<<dim3((unsigned)((n + 255) / 256)), 256, 0, stream>>>(x, xb, n);
    }
    {
        unsigned g = (unsigned)(((size_t)D_ * D_ + 255) / 256);
        transpose_cvt_kernel<<<g, 256, 0, stream>>>(Wq,    wqT, D_, D_, D_);
        transpose_cvt_kernel<<<g, 256, 0, stream>>>(Wk,    wkT, D_, D_, D_);
        transpose_cvt_kernel<<<g, 256, 0, stream>>>(Wv,    wvT, D_, D_, D_);
        transpose_cvt_kernel<<<g, 256, 0, stream>>>(Wskip, wsT, D_, D_, D_);
        unsigned gf = (unsigned)(((size_t)CP_ * D_ + 255) / 256);
        transpose_cvt_kernel<<<gf, 256, 0, stream>>>(Wfc, wfcT, D_, C_, CP_);
    }

    // --- stage 2: projections (q, k bf16; v transposed bf16; x_r fp32) ---
    {
        dim3 grid(N_ / 16, D_ / 64);
        proj_gemm_kernel<0><<<grid, 32, 0, stream>>>(xb, wqT, bq,    (void*)qb);
        proj_gemm_kernel<0><<<grid, 32, 0, stream>>>(xb, wkT, bk,    (void*)kb);
        proj_gemm_kernel<1><<<grid, 32, 0, stream>>>(xb, wvT, bv,    (void*)vTb);
        proj_gemm_kernel<2><<<grid, 32, 0, stream>>>(xb, wsT, bskip, (void*)xr);
    }

    // --- stage 3: flash attention ---
    flash_attn_kernel<<<N_ / 16, 256, 0, stream>>>(qb, kb, vTb, av);

    // --- stage 4: gate + mix ---
    beta_mix_kernel<<<N_, 256, 0, stream>>>(av, xr, Wbeta, bbeta, mix);

    // --- stage 5: classifier GEMM ---
    {
        dim3 grid(N_ / 16, CP_ / 64);
        fc_gemm_kernel<<<grid, 32, 0, stream>>>(mix, wfcT, bfc, out);
    }
}
